// MaskedAttentionHead_57578331570829
// MI455X (gfx1250) — matrix operand order repaired and checked by reading, not compiled, not run on hardware
//
#include <hip/hip_runtime.h>
#include <math.h>

// MI455X / gfx1250: wave32, WMMA 16x16x32 f16 -> f32 accumulate.

typedef __attribute__((ext_vector_type(16))) _Float16 v16h;
typedef __attribute__((ext_vector_type(8)))  float    v8f;

#define B_  4
#define T_  4096
#define C_  1024
#define H_  64
#define BT_ (B_ * T_)            // 16384 rows
#define ROWTILES (BT_ / 16)      // 1024
#define QTILES (B_ * (T_ / 16))  // 1024
#define NSEG 4                   // split-KV factor for flash attention
#define WCHUNK 6144              // f16 elements of one kk-chunk of wfrag (12KB)

__device__ __forceinline__ v8f wmma16(v16h a, v16h b, v8f c) {
  // (neg_a, A, neg_b, B, c_mod, C, reuse_a, reuse_b)
  return __builtin_amdgcn_wmma_f32_16x16x32_f16(false, a, false, b, (short)0, c,
                                                false, false);
}

// Load 16 contiguous f16 (32 bytes, 16B-aligned) into a v16h (global or LDS).
__device__ __forceinline__ v16h ldv16h(const _Float16* p) {
  union { float4 f[2]; v16h h; } u;
  const float4* q = (const float4*)p;
  u.f[0] = q[0];
  u.f[1] = q[1];
  return u.h;
}

// Row reductions across the 16 lanes of a half-wave (xor masks 1,2,4,8 never
// cross the 16-lane boundary, so the two C-layout halves stay independent).
__device__ __forceinline__ float rowmax16(float v) {
#pragma unroll
  for (int m = 1; m < 16; m <<= 1) v = fmaxf(v, __shfl_xor(v, m, 32));
  return v;
}
__device__ __forceinline__ float rowsum16(float v) {
#pragma unroll
  for (int m = 1; m < 16; m <<= 1) v += __shfl_xor(v, m, 32);
  return v;
}

// ---------------------------------------------------------------------------
// Kernel 1: pre-swizzle W{q,k,v} (fp32, C x H) into WMMA B-fragment order f16,
// kk-major so each kk-chunk (12 fragments = 6144 f16 = 12KB) is contiguous:
//   offset = kk*6144 + g*512 + lane*16 + e,   g = w*4 + n
//   value  = W_w[32*kk + (e&7) + 8*(lane>=16) + 16*(e>>3)][16*n + lane%16]
// Total 196608 elements; grid 768 x 256 exactly covers it.
// ---------------------------------------------------------------------------
__global__ void prep_wfrag(const float* __restrict__ Wq,
                           const float* __restrict__ Wk,
                           const float* __restrict__ Wv,
                           _Float16* __restrict__ wfrag) {
  const int idx  = blockIdx.x * blockDim.x + threadIdx.x;
  const int e    = idx & 15;
  const int lane = (idx >> 4) & 31;
  const int t    = idx >> 9;     // kk*12 + g, 0..383
  const int g    = t % 12;
  const int kk   = t / 12;
  const int w    = g >> 2;
  const int n    = g & 3;
  const int half = lane >> 4;
  const int ln   = lane & 15;
  const int krow = 32 * kk + (e & 7) + 8 * half + 16 * (e >> 3);   // B-fragment K index
  const int col  = 16 * n + ln;               // B-fragment N index
  const float* W = (w == 0) ? Wq : (w == 1) ? Wk : Wv;
  wfrag[idx] = (_Float16)W[(size_t)krow * H_ + col];
}

// ---------------------------------------------------------------------------
// Kernel 2: fused QKV projection. One wave per 16-row tile of x; the 8 waves
// of a block share identical W fragments through a double-buffered 2x12KB LDS
// pipeline with DISTANCE-2 register prefetch: chunk kk+2 is loaded from global
// at iteration kk (ping-pong register sets), and ds-stored at iteration kk+1,
// so the load->store wait is covered by a full iteration of 12 WMMAs.
// __launch_bounds__(256, 1): register-heavy kernel; keep the staging sets in
// VGPRs instead of spilling to scratch (observed with the default budget).
// Q, K stored row-major f16 [BT][64]; V stored transposed f16 [B][64][T].
// ---------------------------------------------------------------------------
__global__ void __launch_bounds__(256, 1)
qkv_gemm_wmma(const float* __restrict__ x,
              const _Float16* __restrict__ wfrag,
              _Float16* __restrict__ Qo,
              _Float16* __restrict__ Ko,
              _Float16* __restrict__ VTo) {
  __shared__ __align__(16) _Float16 wlds[2][WCHUNK];  // 24KB

  const int tid  = threadIdx.x;
  const int lane = tid & 31;
  const int wid  = tid >> 5;
  const int rt   = blockIdx.x * 8 + wid;   // row tile 0..1023
  const int row0 = rt * 16;
  const int half = lane >> 4;
  const int ln   = lane & 15;

  v8f acc[3][4];
#pragma unroll
  for (int w = 0; w < 3; ++w)
#pragma unroll
    for (int n = 0; n < 4; ++n)
#pragma unroll
      for (int r = 0; r < 8; ++r) acc[w][n][r] = 0.0f;

  const float* xrow = x + (size_t)(row0 + ln) * C_;  // A lane -> row M=ln

  // --- staging: two named register sets (chunk c lives in set c&1) --------
  float4 wA0, wA1, wA2;  // set A
  float4 wB0, wB1, wB2;  // set B
  union { float4 f[4]; float s[16]; } areg;  // 16 fp32 of the A fragment

  const float4* wsrc0 = (const float4*)(wfrag + (size_t)tid * 24);
  const size_t wstep  = WCHUNK / 8;  // float4 stride per chunk (768)

  auto load_a = [&](int kk) {
    const float* base = xrow + 32 * kk + 8 * half;
    areg.f[0] = ((const float4*)base)[0];
    areg.f[1] = ((const float4*)base)[1];
    areg.f[2] = ((const float4*)(base + 16))[0];
    areg.f[3] = ((const float4*)(base + 16))[1];
  };

  // --- prologue: chunk0 -> set A -> lds[0]; chunk1 -> set B (in flight) ---
  load_a(0);
  {
    const float4* s = wsrc0;                // chunk 0
    wA0 = s[0]; wA1 = s[1]; wA2 = s[2];
  }
  {
    float4* d = (float4*)(&wlds[0][tid * 24]);
    d[0] = wA0; d[1] = wA1; d[2] = wA2;
  }
  {
    const float4* s = wsrc0 + wstep;        // chunk 1
    wB0 = s[0]; wB1 = s[1]; wB2 = s[2];
  }
  __syncthreads();

  for (int kk = 0; kk < 32; ++kk) {
    const int buf = kk & 1;

    // Convert this iteration's A fragment, then issue next A load (HBM) and
    // the distance-2 W load (chunk kk+2 into set (kk&1): A when kk even).
    v16h a;
#pragma unroll
    for (int i = 0; i < 16; ++i) a[i] = (_Float16)areg.s[i];
    if (kk < 31) load_a(kk + 1);
    if (kk < 30) {
      const float4* s = wsrc0 + (size_t)(kk + 2) * wstep;
      if ((kk & 1) == 0) { wA0 = s[0]; wA1 = s[1]; wA2 = s[2]; }
      else               { wB0 = s[0]; wB1 = s[1]; wB2 = s[2]; }
    }

    // 12 WMMAs fed from LDS (fragment g = w*4+n lives at g*512 + lane*16).
    const _Float16* wl = &wlds[buf][lane * 16];
#pragma unroll
    for (int n = 0; n < 4; ++n) {
#pragma unroll
      for (int w = 0; w < 3; ++w) {
        v16h b = ldv16h(wl + (w * 4 + n) * 512);
        acc[w][n] = wmma16(a, b, acc[w][n]);
      }
    }

    // Stage chunk kk+1 (set (kk+1)&1: B when kk even) into the other buffer;
    // its global load was issued one full iteration ago.
    if (kk < 31) {
      float4* d = (float4*)(&wlds[buf ^ 1][tid * 24]);
      if ((kk & 1) == 0) { d[0] = wB0; d[1] = wB1; d[2] = wB2; }
      else               { d[0] = wA0; d[1] = wA1; d[2] = wA2; }
    }
    __syncthreads();
  }

  // C/D layout: element r of acc -> row = row0 + r + 8*half, col = 16n + ln
#pragma unroll
  for (int r = 0; r < 8; ++r) {
    const int row = row0 + r + 8 * half;
    const int bb  = row >> 12;         // row / 4096
    const int tt  = row & (T_ - 1);
#pragma unroll
    for (int n = 0; n < 4; ++n) {
      const int h = 16 * n + ln;
      Qo[(size_t)row * H_ + h] = (_Float16)acc[0][n][r];
      Ko[(size_t)row * H_ + h] = (_Float16)acc[1][n][r];
      VTo[((size_t)bb * H_ + h) * T_ + tt] = (_Float16)acc[2][n][r];
    }
  }
}

// ---------------------------------------------------------------------------
// Kernel 3: causal flash attention, SPLIT-KV: 4 waves per (batch, 16-query
// tile), each owning a disjoint range of 32-key blocks. Stores un-normalized
// partial O plus per-row (m, l) for the merge kernel. 4096 waves total.
// ---------------------------------------------------------------------------
__global__ void flash_attn_split(const _Float16* __restrict__ Qw,
                                 const _Float16* __restrict__ Kw,
                                 const _Float16* __restrict__ VTw,
                                 float* __restrict__ Opart,
                                 float* __restrict__ Mpart,
                                 float* __restrict__ Lpart) {
  __shared__ __align__(16) _Float16 plds[8][16][32];  // per-wave P tile (8KB)

  const int lane = threadIdx.x & 31;
  const int wid  = threadIdx.x >> 5;
  const int tile = blockIdx.x * 8 + wid;  // 0..4095 = seg*1024 + (b*256 + qt)
  const int seg  = tile >> 10;
  const int bq   = tile & 1023;
  const int b    = bq >> 8;
  const int qt   = bq & 255;
  const int q0   = qt * 16;
  const int half = lane >> 4;
  const int ln   = lane & 15;

  const _Float16* Qb = Qw + (size_t)b * T_ * H_;
  const _Float16* Kb = Kw + (size_t)b * T_ * H_;
  const _Float16* Vb = VTw + (size_t)b * H_ * T_;  // [64][4096]

  // Q A-fragments: chunk c covers h in [32c, 32c+32)
  v16h aQ[2];
#pragma unroll
  for (int c = 0; c < 2; ++c) {
    const _Float16* p = Qb + (size_t)(q0 + ln) * H_ + 32 * c + 16 * half;
    union { float4 f[2]; v16h h; } u;
    u.f[0] = ((const float4*)p)[0];
    u.f[1] = ((const float4*)(p + 8))[0];
    aQ[c] = u.h;
  }

  float m[8], l[8];
  v8f o[4];
#pragma unroll
  for (int r = 0; r < 8; ++r) { m[r] = -INFINITY; l[r] = 0.0f; }
#pragma unroll
  for (int n = 0; n < 4; ++n)
#pragma unroll
    for (int r = 0; r < 8; ++r) o[n][r] = 0.0f;

  const int nb = qt / 2 + 1;  // 32-key blocks covering keys <= q0+15
  const int j0 = (seg * nb) >> 2;
  const int j1 = ((seg + 1) * nb) >> 2;

  for (int j = j0; j < j1; ++j) {
    const int k0 = 32 * j;

    // K^T B-fragments + S = Q K^T (per 16-key subtile)
    float st[2][8];
#pragma unroll
    for (int t = 0; t < 2; ++t) {
      const int kb = k0 + 16 * t;
      v8f s;
#pragma unroll
      for (int r = 0; r < 8; ++r) s[r] = 0.0f;
      v16h b0 = ldv16h(Kb + (size_t)(kb + ln) * H_ + 16 * half);
      v16h b1 = ldv16h(Kb + (size_t)(kb + ln) * H_ + 32 + 16 * half);
      s = wmma16(aQ[0], b0, s);
      s = wmma16(aQ[1], b1, s);
      const int kg = kb + ln;  // this lane's key column
#pragma unroll
      for (int r = 0; r < 8; ++r) {
        const int qg = q0 + r + 8 * half;       // element's query row
        const float sv = s[r] * 0.125f;         // scale = H^-0.5
        st[t][r] = (kg > qg) ? -INFINITY : sv;  // causal mask (cndmask)
      }
    }

    // V B-fragments: issue early so they are in flight during the softmax.
    v16h bv[4];
#pragma unroll
    for (int n = 0; n < 4; ++n)
      bv[n] = ldv16h(Vb + (size_t)(16 * n + ln) * T_ + k0 + 16 * half);

    // Online softmax; stats index r maps to row q0 + r + 8*half per lane-half.
    float fs[8];
#pragma unroll
    for (int r = 0; r < 8; ++r) {
      float v = rowmax16(fmaxf(st[0][r], st[1][r]));
      const float mn = fmaxf(m[r], v);
      fs[r] = __expf(m[r] - mn);
      const float p0 = __expf(st[0][r] - mn);
      const float p1 = __expf(st[1][r] - mn);
      plds[wid][r + 8 * half][ln]      = (_Float16)p0;
      plds[wid][r + 8 * half][16 + ln] = (_Float16)p1;
      l[r] = l[r] * fs[r] + rowsum16(p0 + p1);
      m[r] = mn;
    }
#pragma unroll
    for (int n = 0; n < 4; ++n)
#pragma unroll
      for (int r = 0; r < 8; ++r) o[n][r] *= fs[r];

    // Re-read P in A-fragment layout (same-wave DS ops are in order).
    union { float4 f[2]; v16h h; } up;
    up.f[0] = *(const float4*)&plds[wid][ln][16 * half];
    up.f[1] = *(const float4*)&plds[wid][ln][16 * half + 8];
    const v16h aP = up.h;

#pragma unroll
    for (int n = 0; n < 4; ++n) o[n] = wmma16(aP, bv[n], o[n]);
  }

  // Store un-normalized partials + per-row stats.
  const size_t pbase = (size_t)tile * (16 * H_);
#pragma unroll
  for (int r = 0; r < 8; ++r) {
    const int rl = r + 8 * half;
#pragma unroll
    for (int n = 0; n < 4; ++n)
      Opart[pbase + (size_t)rl * H_ + 16 * n + ln] = o[n][r];
    if (ln == 0) {
      Mpart[(size_t)tile * 16 + rl] = m[r];
      Lpart[(size_t)tile * 16 + rl] = l[r];
    }
  }
}

// ---------------------------------------------------------------------------
// Kernel 4: log-sum-exp merge of the NSEG partials. One thread per output
// element (B*T*H = 1,048,576 threads).
// ---------------------------------------------------------------------------
__global__ void flash_merge(const float* __restrict__ Opart,
                            const float* __restrict__ Mpart,
                            const float* __restrict__ Lpart,
                            float* __restrict__ out) {
  const int idx = blockIdx.x * blockDim.x + threadIdx.x;  // b*T*H + q*H + h
  const int h  = idx & (H_ - 1);
  const int q  = (idx >> 6) & (T_ - 1);
  const int b  = idx >> 18;
  const int bq = b * 256 + (q >> 4);  // tile within a segment
  const int rl = q & 15;

  float M = -INFINITY;
#pragma unroll
  for (int s = 0; s < NSEG; ++s)
    M = fmaxf(M, Mpart[(size_t)(s * 1024 + bq) * 16 + rl]);

  float num = 0.0f, den = 0.0f;
#pragma unroll
  for (int s = 0; s < NSEG; ++s) {
    const int part = s * 1024 + bq;
    const float w = __expf(Mpart[(size_t)part * 16 + rl] - M);
    den += w * Lpart[(size_t)part * 16 + rl];
    num += w * Opart[(size_t)part * (16 * H_) + (size_t)rl * H_ + h];
  }
  out[idx] = num / den;
}

// ---------------------------------------------------------------------------
extern "C" void kernel_launch(void* const* d_in, const int* in_sizes, int n_in,
                              void* d_out, int out_size, void* d_ws,
                              size_t ws_size, hipStream_t stream) {
  (void)in_sizes; (void)n_in; (void)out_size; (void)ws_size;
  const float* x  = (const float*)d_in[0];
  const float* Wq = (const float*)d_in[1];
  const float* Wk = (const float*)d_in[2];
  const float* Wv = (const float*)d_in[3];
  float* out = (float*)d_out;

  // Workspace layout:
  //   f16:  Q[BT*64] | K[BT*64] | V^T[BT*64] | Wfrag[196608]   (~6.7 MB)
  //   f32:  Opart[4096*1024] | Mpart[65536] | Lpart[65536]     (~17.3 MB)
  _Float16* ws = (_Float16*)d_ws;
  _Float16* Qw = ws;
  _Float16* Kw = ws + (size_t)1 * BT_ * H_;
  _Float16* VT = ws + (size_t)2 * BT_ * H_;
  _Float16* wf = ws + (size_t)3 * BT_ * H_;
  float* fws   = (float*)(ws + (size_t)3 * BT_ * H_ + 196608);
  float* Opart = fws;                                   // 4096 * 1024
  float* Mpart = fws + (size_t)NSEG * QTILES * 1024;    // 4096 * 16
  float* Lpart = Mpart + (size_t)NSEG * QTILES * 16;

  prep_wfrag<<<768, 256, 0, stream>>>(Wq, Wk, Wv, wf);
  qkv_gemm_wmma<<<ROWTILES / 8, 256, 0, stream>>>(x, wf, Qw, Kw, VT);
  flash_attn_split<<<(NSEG * QTILES) / 8, 256, 0, stream>>>(Qw, Kw, VT, Opart,
                                                            Mpart, Lpart);
  flash_merge<<<(BT_ * H_) / 256, 256, 0, stream>>>(Opart, Mpart, Lpart, out);
}
